// ConditionalNVPFlow_24970939859619
// MI455X (gfx1250) — matrix operand order repaired and checked by reading, not compiled, not run on hardware
//
#include <hip/hip_runtime.h>
#include <hip/hip_bf16.h>

// ConditionalNVPFlow for MI455X (gfx1250).
// B=1024, S=64, D_Y=64, D_X=8, H=64, L=4, NP=5256.
// d_out = [ x (1024*64*8 f32) | log_probs (1024*64 f32) ]
// d_ws (floats):
//   [0 .. 10764288)          Pl   (1024x10512 f32; first 1M floats alias t0,t1)
//   [10764288 .. 13455360)   hWb  (10512x512 bf16 pairs, u32)
//   [13455360 .. 13488128)   yb
//   [13488128 .. 13504512)   W0b
//   [13504512 .. 13635584)   W1b
//   [13635584 .. 13897728)   t0b
//   [13897728 .. 14159872)   t1b      total ~56.6 MB

typedef __attribute__((ext_vector_type(2)))  float        v2f;
typedef __attribute__((ext_vector_type(8)))  float        v8f;
typedef __attribute__((ext_vector_type(8)))  unsigned int v8u;
typedef __attribute__((ext_vector_type(16))) __bf16       v16bf;

__device__ __forceinline__ unsigned int pack_bf16(float lo, float hi) {
    unsigned int a = __float_as_uint(lo);
    unsigned int b = __float_as_uint(hi);
    a = (a + 0x7FFFu + ((a >> 16) & 1u)) >> 16;   // round-to-nearest-even
    b = (b + 0x7FFFu + ((b >> 16) & 1u)) >> 16;
    return (b << 16) | (a & 0xFFFFu);
}

// ---------------------------------------------------------------------------
// init: x <- z ; lp <- -0.5*sum(z^2) - 0.5*D_X*log(2*pi)
// ---------------------------------------------------------------------------
__global__ __launch_bounds__(256) void init_base(const float* __restrict__ z,
                                                 float* __restrict__ x,
                                                 float* __restrict__ lp) {
    int idx = blockIdx.x * 256 + threadIdx.x;           // 0..65535
    const float4* zp = (const float4*)(z + (size_t)idx * 8);
    float4 a = zp[0];
    float4 b = zp[1];
    float ss = a.x * a.x + a.y * a.y + a.z * a.z + a.w * a.w
             + b.x * b.x + b.y * b.y + b.z * b.z + b.w * b.w;
    float4* xp = (float4*)(x + (size_t)idx * 8);
    xp[0] = a;
    xp[1] = b;
    lp[idx] = -0.5f * ss - 7.3515082656373813f;         // 4*log(2*pi)
}

// ---------------------------------------------------------------------------
// pack f32 -> packed bf16 pairs (u32). One float4 -> uint2 per thread.
// ---------------------------------------------------------------------------
__global__ __launch_bounds__(256) void pack_quads(const float* __restrict__ in,
                                                  unsigned int* __restrict__ out,
                                                  int nquads) {
    int i = blockIdx.x * 256 + threadIdx.x;
    if (i < nquads) {
        float4 f = ((const float4*)in)[i];
        uint2 o;
        o.x = pack_bf16(f.x, f.y);
        o.y = pack_bf16(f.z, f.w);
        ((uint2*)out)[i] = o;
    }
}

// ---------------------------------------------------------------------------
// bf16-WMMA GEMM:  C[M,N] = act( A[M,K] * Bw[N,K]^T + bias[N] )
// A,Bw are PRE-PACKED bf16 (u32 pairs, row stride K/2). Tiles 128x128x32,
// 8 waves, each wave 4x2 16x16 tiles. Double-buffered LDS staging via CDNA5
// async global->LDS (global_load_async_to_lds_b128, ASYNCcnt): tile i+1 is
// fetched while tile i is consumed by WMMA.
// M % 128 == 0, K % 32 == 0; N handled with guards (OOB B rows pre-zeroed
// once, outside the pipeline).
// ---------------------------------------------------------------------------
__global__ __launch_bounds__(256) void gemm_bf16(const unsigned int* __restrict__ Ab,
                                                 const unsigned int* __restrict__ Bb,
                                                 const float* __restrict__ bias,
                                                 float* __restrict__ C,
                                                 int M, int N, int K, int act) {
    __shared__ unsigned int As[2][128][16];   // bf16 pairs: As[buf][row][k/2]
    __shared__ unsigned int Bs[2][128][16];   // bf16 pairs: Bs[buf][col][k/2]

    const int tid  = threadIdx.x;
    const int lane = tid & 31;
    const int wid  = tid >> 5;
    const int half = lane >> 4;
    const int l15  = lane & 15;
    const int bm   = blockIdx.y * 128;
    const int bn   = blockIdx.x * 128;
    const int Kw   = K >> 1;                  // row stride in u32

    const int rt0 = (wid & 1) * 4;
    const int ct0 = (wid >> 1) * 2;

    // staging decomposition: 512 x 16B units per tile, 2 per thread
    const int row0 = tid >> 2;                // unit tid
    const int row1 = (tid + 256) >> 2;        // unit tid+256
    const int chk  = tid & 3;                 // same for both units
    const bool bv0 = (bn + row0) < N;
    const bool bv1 = (bn + row1) < N;

    const unsigned asBase = (unsigned)(unsigned long long)(&As[0][0][0]);
    const unsigned bsBase = (unsigned)(unsigned long long)(&Bs[0][0][0]);

    // ---- one-time zero fill of OOB B rows (constant across k) ----
    if (bn + 127 >= N) {
        float4 zz = {0.f, 0.f, 0.f, 0.f};
        if (!bv0) {
            *(float4*)((char*)&Bs[0][0][0] + (row0 * 64 + chk * 16)) = zz;
            *(float4*)((char*)&Bs[1][0][0] + (row0 * 64 + chk * 16)) = zz;
        }
        if (!bv1) {
            *(float4*)((char*)&Bs[0][0][0] + (row1 * 64 + chk * 16)) = zz;
            *(float4*)((char*)&Bs[1][0][0] + (row1 * 64 + chk * 16)) = zz;
        }
    }

    // ---- async stage of one k-window into LDS buffer p ----
    auto stage = [&](int kw, int p) {
        const unsigned aB = asBase + (unsigned)p * 8192u;
        const unsigned bB = bsBase + (unsigned)p * 8192u;
        unsigned long long ga0 =
            (unsigned long long)(Ab + (size_t)(bm + row0) * Kw + kw + chk * 4);
        unsigned la0 = aB + row0 * 64 + chk * 16;
        asm volatile("global_load_async_to_lds_b128 %0, %1, off"
                     :: "v"(la0), "v"(ga0) : "memory");
        unsigned long long ga1 =
            (unsigned long long)(Ab + (size_t)(bm + row1) * Kw + kw + chk * 4);
        unsigned la1 = aB + row1 * 64 + chk * 16;
        asm volatile("global_load_async_to_lds_b128 %0, %1, off"
                     :: "v"(la1), "v"(ga1) : "memory");
        if (bv0) {
            unsigned long long gb =
                (unsigned long long)(Bb + (size_t)(bn + row0) * Kw + kw + chk * 4);
            unsigned lb = bB + row0 * 64 + chk * 16;
            asm volatile("global_load_async_to_lds_b128 %0, %1, off"
                         :: "v"(lb), "v"(gb) : "memory");
        }
        if (bv1) {
            unsigned long long gb =
                (unsigned long long)(Bb + (size_t)(bn + row1) * Kw + kw + chk * 4);
            unsigned lb = bB + row1 * 64 + chk * 16;
            asm volatile("global_load_async_to_lds_b128 %0, %1, off"
                         :: "v"(lb), "v"(gb) : "memory");
        }
    };

    v8f acc[4][2];
#pragma unroll
    for (int i = 0; i < 4; i++)
#pragma unroll
        for (int j = 0; j < 2; j++)
            acc[i][j] = (v8f){0.f, 0.f, 0.f, 0.f, 0.f, 0.f, 0.f, 0.f};

    const int nk = K >> 5;
    stage(0, 0);                               // prologue: tile 0 -> buf 0

    for (int it = 0; it < nk; it++) {
        const int p = it & 1;
        asm volatile("s_wait_asynccnt 0" ::: "memory");  // my tile-it asyncs done
        __syncthreads();                                  // everyone's done; buf p full
        if (it + 1 < nk) stage((it + 1) << 4, p ^ 1);     // prefetch next tile

        // ---- fragment loads from buffer p ----
        // A 16x32 bf16: lane holds row (l15), pairIdx = (v&3) + half*4 + (v>=4)*8
        v8u au[4];
#pragma unroll
        for (int rt = 0; rt < 4; rt++) {
            int r = (rt0 + rt) * 16 + l15;
#pragma unroll
            for (int v = 0; v < 8; v++)
                au[rt][v] = As[p][r][(v & 3) + (half << 2) + ((v >> 2) << 3)];
        }
        // B 32x16 bf16: lane holds col (l15), pairIdx = (v&3) + half*4 + (v>=4)*8
        v8u bu[2];
#pragma unroll
        for (int ct = 0; ct < 2; ct++) {
            int c = (ct0 + ct) * 16 + l15;
#pragma unroll
            for (int v = 0; v < 8; v++)
                bu[ct][v] = Bs[p][c][(v & 3) + (half << 2) + ((v >> 2) << 3)];
        }

#pragma unroll
        for (int rt = 0; rt < 4; rt++) {
            v16bf af = __builtin_bit_cast(v16bf, au[rt]);
#pragma unroll
            for (int ct = 0; ct < 2; ct++) {
                v16bf bf = __builtin_bit_cast(v16bf, bu[ct]);
                acc[rt][ct] = __builtin_amdgcn_wmma_f32_16x16x32_bf16(
                    false, af, false, bf, (short)0, acc[rt][ct], false, false);
            }
        }
    }

    // ---- epilogue: bias + optional relu, guarded store ----
#pragma unroll
    for (int rt = 0; rt < 4; rt++) {
#pragma unroll
        for (int ct = 0; ct < 2; ct++) {
            int gn = bn + (ct0 + ct) * 16 + l15;
            if (gn < N) {
                float bv = bias[gn];
#pragma unroll
                for (int i = 0; i < 8; i++) {
                    int gm = bm + (rt0 + rt) * 16 + (half << 3) + i;
                    float v = acc[rt][ct][i] + bv;
                    if (act) v = fmaxf(v, 0.f);
                    C[(size_t)gm * (size_t)N + gn] = v;
                }
            }
        }
    }
}

// ---------------------------------------------------------------------------
// Coupling kernel: one block per batch sample b. s-net and t-net 3-layer MLPs
// via V_WMMA_F32_16X16X4_F32 (full f32), then affine coupling + logdet update.
// Param layout per net (5256 f32): Wa[64][8] | ba[64] | Wb[64][64] | bb[64]
//                                  | Wc[8][64] | bc[8]
// ---------------------------------------------------------------------------
#define WA_OFF 0
#define BA_OFF 512
#define WB_OFF 576
#define BB_OFF 4672
#define WC_OFF 4736
#define BC_OFF 5248

__global__ __launch_bounds__(256) void coupling(const float* __restrict__ Pl,
                                                const int* __restrict__ maskp,
                                                float* __restrict__ x,
                                                float* __restrict__ lp) {
    __shared__ float xs[64][8];
    __shared__ float xm[64][8];
    __shared__ float prm[5256];
    __shared__ float hb[2][64][64];
    __shared__ float st[2][64][8];
    __shared__ float mk[8];

    const int tid  = threadIdx.x;
    const int lane = tid & 31;
    const int wid  = tid >> 5;
    const int half = lane >> 4;
    const int l15  = lane & 15;
    const int b    = blockIdx.x;

    if (tid < 8) mk[tid] = (float)maskp[tid];
    {
        float2 v = ((const float2*)(x + (size_t)b * 512))[tid];
        ((float2*)&xs[0][0])[tid] = v;
    }
    __syncthreads();
    {
        int e = tid * 2;
        xm[e >> 3][e & 7] = xs[e >> 3][e & 7] * mk[e & 7];
        e++;
        xm[e >> 3][e & 7] = xs[e >> 3][e & 7] * mk[e & 7];
    }
    __syncthreads();

    for (int j = 0; j < 2; j++) {   // j=0: s-net, j=1: t-net
        const float* pp = Pl + (size_t)b * 10512 + (size_t)j * 5256;
        for (int i = tid; i < 5256; i += 256) prm[i] = pp[i];
        __syncthreads();

        // layer 1: h1 = tanh(xm @ Wa^T + ba)  (M=64,N=64,K=8)
        for (int t = wid * 2; t < wid * 2 + 2; t++) {
            int tr = t >> 2, tc = t & 3;
            int r = tr * 16 + l15;
            int c = tc * 16 + l15;
            v8f acc = (v8f){0.f, 0.f, 0.f, 0.f, 0.f, 0.f, 0.f, 0.f};
#pragma unroll
            for (int k0 = 0; k0 < 8; k0 += 4) {
                int kk = k0 + half * 2;
                v2f a  = (v2f){xm[r][kk], xm[r][kk + 1]};
                v2f bw = (v2f){prm[WA_OFF + c * 8 + kk], prm[WA_OFF + c * 8 + kk + 1]};
                acc = __builtin_amdgcn_wmma_f32_16x16x4_f32(
                    false, a, false, bw, (short)0, acc, false, false);
            }
#pragma unroll
            for (int i = 0; i < 8; i++) {
                int m_ = tr * 16 + half * 8 + i;
                hb[0][m_][c] = tanhf(acc[i] + prm[BA_OFF + c]);
            }
        }
        __syncthreads();

        // layer 2: h2 = tanh(h1 @ Wb^T + bb)  (M=64,N=64,K=64)
        for (int t = wid * 2; t < wid * 2 + 2; t++) {
            int tr = t >> 2, tc = t & 3;
            int r = tr * 16 + l15;
            int c = tc * 16 + l15;
            v8f acc = (v8f){0.f, 0.f, 0.f, 0.f, 0.f, 0.f, 0.f, 0.f};
#pragma unroll
            for (int k0 = 0; k0 < 64; k0 += 4) {
                int kk = k0 + half * 2;
                v2f a  = (v2f){hb[0][r][kk], hb[0][r][kk + 1]};
                v2f bw = (v2f){prm[WB_OFF + c * 64 + kk], prm[WB_OFF + c * 64 + kk + 1]};
                acc = __builtin_amdgcn_wmma_f32_16x16x4_f32(
                    false, a, false, bw, (short)0, acc, false, false);
            }
#pragma unroll
            for (int i = 0; i < 8; i++) {
                int m_ = tr * 16 + half * 8 + i;
                hb[1][m_][c] = tanhf(acc[i] + prm[BB_OFF + c]);
            }
        }
        __syncthreads();

        // layer 3: out = h2 @ Wc^T + bc  (M=64,N=8 pad 16,K=64)
        if (wid < 4) {
            int tr = wid;
            int r = tr * 16 + l15;
            int c = l15;
            bool cv = (c < 8);
            v8f acc = (v8f){0.f, 0.f, 0.f, 0.f, 0.f, 0.f, 0.f, 0.f};
#pragma unroll
            for (int k0 = 0; k0 < 64; k0 += 4) {
                int kk = k0 + half * 2;
                v2f a  = (v2f){hb[1][r][kk], hb[1][r][kk + 1]};
                v2f bw = cv ? (v2f){prm[WC_OFF + c * 64 + kk], prm[WC_OFF + c * 64 + kk + 1]}
                            : (v2f){0.f, 0.f};
                acc = __builtin_amdgcn_wmma_f32_16x16x4_f32(
                    false, a, false, bw, (short)0, acc, false, false);
            }
            if (cv) {
#pragma unroll
                for (int i = 0; i < 8; i++) {
                    int m_ = tr * 16 + half * 8 + i;
                    st[j][m_][c] = acc[i] + prm[BC_OFF + c];
                }
            }
        }
        __syncthreads();   // also protects prm reuse for j=1
    }

    // ---- affine coupling update + logdet ----
    {
        float* xp = x + (size_t)b * 512;
        int e0 = tid * 2;
#pragma unroll
        for (int e = e0; e < e0 + 2; e++) {
            int s_ = e >> 3, i_ = e & 7;
            float m  = mk[i_];
            float xv = xs[s_][i_];
            float sv = st[0][s_][i_];
            float tv = st[1][s_][i_];
            float xn = m * xv + (1.f - m) * (xv * expf(sv) + tv);
            xp[e] = xn;
        }
        if (tid < 64) {
            float sum = 0.f;
#pragma unroll
            for (int i = 0; i < 8; i++) sum += (1.f - mk[i]) * st[0][tid][i];
            lp[(size_t)b * 64 + tid] -= sum;
        }
    }
}

// ---------------------------------------------------------------------------
// Host launcher
// inputs: 0:y 1:z 2:W0 3:b0 4:W1 5:b1 6:head_W 7:head_b 8:masks(int32)
// ---------------------------------------------------------------------------
extern "C" void kernel_launch(void* const* d_in, const int* in_sizes, int n_in,
                              void* d_out, int out_size, void* d_ws, size_t ws_size,
                              hipStream_t stream) {
    const float* y     = (const float*)d_in[0];
    const float* z     = (const float*)d_in[1];
    const float* W0    = (const float*)d_in[2];
    const float* b0    = (const float*)d_in[3];
    const float* W1    = (const float*)d_in[4];
    const float* b1    = (const float*)d_in[5];
    const float* headW = (const float*)d_in[6];
    const float* headb = (const float*)d_in[7];
    const int*   masks = (const int*)d_in[8];

    float* out   = (float*)d_out;
    float* xout  = out;             // 1024*64*8 = 524288 floats
    float* lpout = out + 524288;    // 1024*64   = 65536 floats

    float* ws = (float*)d_ws;
    float*        Pl  = ws;                                   // 1024*10512 f32
    float*        t0  = Pl;                                   // alias (dies early)
    float*        t1  = Pl + 524288;                          // alias (dies early)
    unsigned int* hWb = (unsigned int*)(ws + 10764288);       // 10512*512/2 u32
    unsigned int* yb  = (unsigned int*)(ws + 13455360);       // 32768 u32
    unsigned int* W0b = (unsigned int*)(ws + 13488128);       // 16384 u32
    unsigned int* W1b = (unsigned int*)(ws + 13504512);       // 131072 u32
    unsigned int* t0b = (unsigned int*)(ws + 13635584);       // 262144 u32
    unsigned int* t1b = (unsigned int*)(ws + 13897728);       // 262144 u32

    // base log-prob + copy z -> x
    init_base<<<256, 256, 0, stream>>>(z, xout, lpout);

    // pack static operands to bf16
    pack_quads<<<(16384 + 255) / 256, 256, 0, stream>>>(y,  yb,  16384);   // 1024*64/4
    pack_quads<<<(8192  + 255) / 256, 256, 0, stream>>>(W0, W0b, 8192);    // 512*64/4
    pack_quads<<<(65536 + 255) / 256, 256, 0, stream>>>(W1, W1b, 65536);   // 512*512/4

    // conditioner: t0 = relu(y @ W0^T + b0); t1 = relu(t0 @ W1^T + b1)
    dim3 gc(512 / 128, 1024 / 128);
    gemm_bf16<<<gc, 256, 0, stream>>>(yb, W0b, b0, t0, 1024, 512, 64, 1);
    pack_quads<<<(131072 + 255) / 256, 256, 0, stream>>>(t0, t0b, 131072); // 1024*512/4
    gemm_bf16<<<gc, 256, 0, stream>>>(t0b, W1b, b1, t1, 1024, 512, 512, 1);
    pack_quads<<<(131072 + 255) / 256, 256, 0, stream>>>(t1, t1b, 131072);

    // per flow layer: pack head slice, project both heads, apply coupling
    dim3 gp((10512 + 127) / 128, 1024 / 128);   // (83, 8)
    const int headQuads = 10512 * 512 / 4;      // 1,345,536
    for (int l = 0; l < 4; l++) {
        pack_quads<<<(headQuads + 255) / 256, 256, 0, stream>>>(
            headW + (size_t)(2 * l) * 5256 * 512, hWb, headQuads);
        gemm_bf16<<<gp, 256, 0, stream>>>(t1b, hWb,
                                          headb + (size_t)(2 * l) * 5256,
                                          Pl, 1024, 10512, 512, 0);
        coupling<<<1024, 256, 0, stream>>>(Pl, masks + l * 8, xout, lpout);
    }
}
